// Decoder_87041807221332
// MI455X (gfx1250) — matrix operand order repaired and checked by reading, not compiled, not run on hardware
//
#include <hip/hip_runtime.h>
#include <hip/hip_bf16.h>

#define B_     1024
#define H_     512
#define STEPS_ 64
#define EXO_   8
#define IN_    9
#define ZD_    16
#define G4H    (4 * H_)

typedef __attribute__((ext_vector_type(16))) __bf16 v16bf;
typedef __attribute__((ext_vector_type(8)))  float  v8f;

// ---------- WMMA fragment loaders (16-bit, 16x16x32 bf16) ----------
// A (16x32, MxK): lane m = lane&15; lanes 0-15 cover K = kb+0..7 (v0..3) and
// kb+16..23 (v4..7); lanes 16-31 cover kb+8..15 and kb+24..31.
// Caller passes p = rowBase + kb + (lane>>4)*8 : chunks at +0 and +16.
__device__ __forceinline__ v16bf load_fragA(const __bf16* p) {
    union { v16bf v; float4 f[2]; } u;
    u.f[0] = *reinterpret_cast<const float4*>(p);
    u.f[1] = *reinterpret_cast<const float4*>(p + 16);
    return u.v;
}
// B (32x16, KxN): lane n = lane&15; same K layout as A: lanes 0-15 hold
// kb+0..7 and kb+16..23, lanes 16-31 kb+8..15 and kb+24..31. Caller passes p = rowBase + kb + (lane>>4)*8.
__device__ __forceinline__ v16bf load_fragB(const __bf16* p) {
    union { v16bf v; float4 f[2]; } u;
    u.f[0] = *reinterpret_cast<const float4*>(p);
    u.f[1] = *reinterpret_cast<const float4*>(p + 16);
    return u.v;
}

__device__ __forceinline__ float sigmoidf_(float x) {
    return 1.0f / (1.0f + __expf(-x));
}

// ---------- preprocessing ----------
__global__ __launch_bounds__(256) void convert_w_k(
    const float* __restrict__ w0, const float* __restrict__ w1,
    const float* __restrict__ w2,
    __bf16* __restrict__ o0, __bf16* __restrict__ o1, __bf16* __restrict__ o2) {
    int i = blockIdx.x * blockDim.x + threadIdx.x;
    if (i < G4H * H_) {
        o0[i] = (__bf16)w0[i];
        o1[i] = (__bf16)w1[i];
        o2[i] = (__bf16)w2[i];
    }
}

__global__ __launch_bounds__(256) void init_state_k(
    const float* __restrict__ h0, const float* __restrict__ c0,
    const float* __restrict__ y0, const float* __restrict__ z,
    const float* __restrict__ Wz, const float* __restrict__ bz,
    __bf16* __restrict__ h0b, __bf16* __restrict__ h1b,
    float* __restrict__ c0s, float* __restrict__ c1s,
    float* __restrict__ prev_y, float* __restrict__ zbias) {
    const int i  = blockIdx.x * blockDim.x + threadIdx.x;
    const int BH = B_ * H_;
    if (i < BH) {
        h0b[i] = (__bf16)h0[i];
        h1b[i] = (__bf16)h0[BH + i];
        c0s[i] = c0[i];
        c1s[i] = c0[BH + i];
    }
    if (i < B_) prev_y[i] = y0[i];
    if (i < B_ * IN_) {
        int b = i / IN_, j = i % IN_;
        float s = bz[j];
#pragma unroll
        for (int k = 0; k < ZD_; ++k) s += z[b * ZD_ + k] * Wz[j * ZD_ + k];
        zbias[i] = s;
    }
}

// ---------- layer 0: gates = in_t@Wih0^T (fp32, K=9) + h@Whh0^T (WMMA bf16)
//            + fused cell. Each wave: 64(M) x 16(N hidden) block, 4 gates,
//            16 v8f accumulators; each B fragment feeds 4 WMMAs. ----------
__global__ __launch_bounds__(256) void lstm_l0_k(
    const __bf16* __restrict__ hA,       // [B,H] prev layer-0 h (bf16)
    const __bf16* __restrict__ Whh,      // [4H,H] bf16
    const float*  __restrict__ Wih,      // [4H,IN] fp32
    const float*  __restrict__ bih, const float* __restrict__ bhh,
    const float*  __restrict__ prev_y,   // [B]
    const float*  __restrict__ x_future, // [B,STEPS,EXO]
    const float*  __restrict__ zbias,    // [B,IN]
    float*        __restrict__ cS,       // [B,H] c state (in/out)
    __bf16*       __restrict__ hOut,     // [B,H] new h (bf16)
    int t) {
    const int lane = threadIdx.x & 31;
    const int wave = threadIdx.x >> 5;
    const int tile = blockIdx.x * 8 + wave;   // 512 tiles: 16 mB x 32 nT
    const int mB = tile >> 5;
    const int nT = tile & 31;
    const int m0 = mB * 64, n0 = nT * 16;
    const int lm = lane & 15;
    const int kh = lane >> 4;

    const v8f z8 = {0.f, 0.f, 0.f, 0.f, 0.f, 0.f, 0.f, 0.f};
    v8f acc[4][4];   // [gate][m-subtile]
#pragma unroll
    for (int g = 0; g < 4; ++g)
#pragma unroll
        for (int ms = 0; ms < 4; ++ms) acc[g][ms] = z8;

    const __bf16* aBase = hA + (size_t)(m0 + lm) * H_ + kh * 8;
#pragma unroll 2
    for (int kk = 0; kk < 16; ++kk) {
        const int kb = kk * 32;
        v16bf a[4];
#pragma unroll
        for (int ms = 0; ms < 4; ++ms)
            a[ms] = load_fragA(aBase + (size_t)ms * 16 * H_ + kb);
#pragma unroll
        for (int g = 0; g < 4; ++g) {
            v16bf b = load_fragB(Whh + (size_t)(g * H_ + n0 + lm) * H_ +
                                 kh * 8 + kb);
#pragma unroll
            for (int ms = 0; ms < 4; ++ms)
                acc[g][ms] = __builtin_amdgcn_wmma_f32_16x16x32_bf16(
                    false, a[ms], false, b, (short)0, acc[g][ms], false, false);
        }
    }

    const int nl = n0 + lm;
    float wi[4][IN_];
    float bsum[4];
#pragma unroll
    for (int g = 0; g < 4; ++g) {
#pragma unroll
        for (int k = 0; k < IN_; ++k) wi[g][k] = Wih[(g * H_ + nl) * IN_ + k];
        bsum[g] = bih[g * H_ + nl] + bhh[g * H_ + nl];
    }

#pragma unroll
    for (int ms = 0; ms < 4; ++ms) {
#pragma unroll
        for (int r = 0; r < 8; ++r) {
            const int m = m0 + ms * 16 + r + kh * 8;  // C/D layout: VGPR r -> M=r(+8)
            float xin[IN_];
            xin[0] = prev_y[m] + zbias[m * IN_ + 0];
#pragma unroll
            for (int e = 0; e < EXO_; ++e)
                xin[1 + e] = x_future[((size_t)m * STEPS_ + t) * EXO_ + e] +
                             zbias[m * IN_ + 1 + e];
            float gv[4];
#pragma unroll
            for (int g = 0; g < 4; ++g) {
                float s = acc[g][ms][r] + bsum[g];
#pragma unroll
                for (int k = 0; k < IN_; ++k) s += xin[k] * wi[g][k];
                gv[g] = s;
            }
            const float cp = cS[(size_t)m * H_ + nl];
            const float ig = sigmoidf_(gv[0]);
            const float fg = sigmoidf_(gv[1]);
            const float gg = tanhf(gv[2]);
            const float og = sigmoidf_(gv[3]);
            const float c2 = fg * cp + ig * gg;
            const float hh = og * tanhf(c2);
            cS[(size_t)m * H_ + nl]   = c2;
            hOut[(size_t)m * H_ + nl] = (__bf16)hh;
        }
    }
}

// ---------- layer 1: gates = h0n@Wih1^T + h1@Whh1^T (both WMMA, two K-loops
//            over shared accumulators) + fused cell ----------
__global__ __launch_bounds__(256) void lstm_l1_k(
    const __bf16* __restrict__ hA0,  // [B,H] this step's layer-0 h
    const __bf16* __restrict__ hA1,  // [B,H] prev layer-1 h
    const __bf16* __restrict__ Wih,  // [4H,H] bf16
    const __bf16* __restrict__ Whh,  // [4H,H] bf16
    const float*  __restrict__ bih, const float* __restrict__ bhh,
    float*        __restrict__ cS,   // [B,H]
    __bf16*       __restrict__ hOut) {
    const int lane = threadIdx.x & 31;
    const int wave = threadIdx.x >> 5;
    const int tile = blockIdx.x * 8 + wave;
    const int mB = tile >> 5;
    const int nT = tile & 31;
    const int m0 = mB * 64, n0 = nT * 16;
    const int lm = lane & 15;
    const int kh = lane >> 4;

    const v8f z8 = {0.f, 0.f, 0.f, 0.f, 0.f, 0.f, 0.f, 0.f};
    v8f acc[4][4];
#pragma unroll
    for (int g = 0; g < 4; ++g)
#pragma unroll
        for (int ms = 0; ms < 4; ++ms) acc[g][ms] = z8;

    // pass 1: h0n @ Wih1^T
    {
        const __bf16* aBase = hA0 + (size_t)(m0 + lm) * H_ + kh * 8;
#pragma unroll 2
        for (int kk = 0; kk < 16; ++kk) {
            const int kb = kk * 32;
            v16bf a[4];
#pragma unroll
            for (int ms = 0; ms < 4; ++ms)
                a[ms] = load_fragA(aBase + (size_t)ms * 16 * H_ + kb);
#pragma unroll
            for (int g = 0; g < 4; ++g) {
                v16bf b = load_fragB(Wih + (size_t)(g * H_ + n0 + lm) * H_ +
                                     kh * 8 + kb);
#pragma unroll
                for (int ms = 0; ms < 4; ++ms)
                    acc[g][ms] = __builtin_amdgcn_wmma_f32_16x16x32_bf16(
                        false, a[ms], false, b, (short)0, acc[g][ms], false,
                        false);
            }
        }
    }
    // pass 2: h1_prev @ Whh1^T
    {
        const __bf16* aBase = hA1 + (size_t)(m0 + lm) * H_ + kh * 8;
#pragma unroll 2
        for (int kk = 0; kk < 16; ++kk) {
            const int kb = kk * 32;
            v16bf a[4];
#pragma unroll
            for (int ms = 0; ms < 4; ++ms)
                a[ms] = load_fragA(aBase + (size_t)ms * 16 * H_ + kb);
#pragma unroll
            for (int g = 0; g < 4; ++g) {
                v16bf b = load_fragB(Whh + (size_t)(g * H_ + n0 + lm) * H_ +
                                     kh * 8 + kb);
#pragma unroll
                for (int ms = 0; ms < 4; ++ms)
                    acc[g][ms] = __builtin_amdgcn_wmma_f32_16x16x32_bf16(
                        false, a[ms], false, b, (short)0, acc[g][ms], false,
                        false);
            }
        }
    }

    const int nl = n0 + lm;
    float bsum[4];
#pragma unroll
    for (int g = 0; g < 4; ++g) bsum[g] = bih[g * H_ + nl] + bhh[g * H_ + nl];

#pragma unroll
    for (int ms = 0; ms < 4; ++ms) {
#pragma unroll
        for (int r = 0; r < 8; ++r) {
            const int m = m0 + ms * 16 + r + kh * 8;
            const float cp = cS[(size_t)m * H_ + nl];
            const float ig = sigmoidf_(acc[0][ms][r] + bsum[0]);
            const float fg = sigmoidf_(acc[1][ms][r] + bsum[1]);
            const float gg = tanhf(acc[2][ms][r] + bsum[2]);
            const float og = sigmoidf_(acc[3][ms][r] + bsum[3]);
            const float c2 = fg * cp + ig * gg;
            const float hh = og * tanhf(c2);
            cS[(size_t)m * H_ + nl]   = c2;
            hOut[(size_t)m * H_ + nl] = (__bf16)hh;
        }
    }
}

// ---------- projection H->1: one wave32 per batch row ----------
__global__ __launch_bounds__(256) void proj_k(
    const __bf16* __restrict__ h1, const float* __restrict__ Wp,
    const float* __restrict__ bp, float* __restrict__ out,
    float* __restrict__ prev_y, int t) {
    const int lane = threadIdx.x & 31;
    const int row  = blockIdx.x * 8 + (threadIdx.x >> 5);
    float s = 0.f;
#pragma unroll
    for (int k = lane; k < H_; k += 32)
        s += (float)h1[(size_t)row * H_ + k] * Wp[k];
#pragma unroll
    for (int off = 16; off > 0; off >>= 1) s += __shfl_xor(s, off, 32);
    if (lane == 0) {
        const float y = s + bp[0];
        out[(size_t)row * STEPS_ + t] = y;
        prev_y[row] = y;
    }
}

extern "C" void kernel_launch(void* const* d_in, const int* in_sizes, int n_in,
                              void* d_out, int out_size, void* d_ws, size_t ws_size,
                              hipStream_t stream) {
    const float* y0       = (const float*)d_in[0];
    const float* x_future = (const float*)d_in[1];
    const float* h0       = (const float*)d_in[2];
    const float* c0       = (const float*)d_in[3];
    const float* z        = (const float*)d_in[4];
    const float* W_ih0    = (const float*)d_in[5];
    const float* W_hh0    = (const float*)d_in[6];
    const float* b_ih0    = (const float*)d_in[7];
    const float* b_hh0    = (const float*)d_in[8];
    const float* W_ih1    = (const float*)d_in[9];
    const float* W_hh1    = (const float*)d_in[10];
    const float* b_ih1    = (const float*)d_in[11];
    const float* b_hh1    = (const float*)d_in[12];
    const float* W_proj   = (const float*)d_in[13];
    const float* b_proj   = (const float*)d_in[14];
    const float* W_z      = (const float*)d_in[15];
    const float* b_z      = (const float*)d_in[16];
    float* out = (float*)d_out;

    // workspace layout (all offsets 256B aligned)
    char* ws = (char*)d_ws;
    size_t off = 0;
    const size_t WSZ  = (size_t)G4H * H_ * sizeof(__bf16);  // 2 MB
    const size_t HSZ  = (size_t)B_ * H_ * sizeof(__bf16);   // 1 MB
    const size_t CSZ  = (size_t)B_ * H_ * sizeof(float);    // 2 MB
    __bf16* whh0b = (__bf16*)(ws + off); off += WSZ;
    __bf16* wih1b = (__bf16*)(ws + off); off += WSZ;
    __bf16* whh1b = (__bf16*)(ws + off); off += WSZ;
    __bf16* h0b[2]; h0b[0] = (__bf16*)(ws + off); off += HSZ;
                    h0b[1] = (__bf16*)(ws + off); off += HSZ;
    __bf16* h1b[2]; h1b[0] = (__bf16*)(ws + off); off += HSZ;
                    h1b[1] = (__bf16*)(ws + off); off += HSZ;
    float* c0s    = (float*)(ws + off); off += CSZ;
    float* c1s    = (float*)(ws + off); off += CSZ;
    float* zbias  = (float*)(ws + off); off += ((size_t)B_ * IN_ * 4 + 255) & ~255ull;
    float* prev_y = (float*)(ws + off); off += 4096;

    // one-time (per call) preprocessing
    convert_w_k<<<(G4H * H_ + 255) / 256, 256, 0, stream>>>(
        W_hh0, W_ih1, W_hh1, whh0b, wih1b, whh1b);
    init_state_k<<<(B_ * H_ + 255) / 256, 256, 0, stream>>>(
        h0, c0, y0, z, W_z, b_z, h0b[0], h1b[0], c0s, c1s, prev_y, zbias);

    // sequential rollout: 3 kernels per step, stream order carries the deps
    for (int t = 0; t < STEPS_; ++t) {
        const int in  = t & 1;
        const int oup = in ^ 1;
        lstm_l0_k<<<64, 256, 0, stream>>>(
            h0b[in], whh0b, W_ih0, b_ih0, b_hh0,
            prev_y, x_future, zbias, c0s, h0b[oup], t);
        lstm_l1_k<<<64, 256, 0, stream>>>(
            h0b[oup], h1b[in], wih1b, whh1b, b_ih1, b_hh1, c1s, h1b[oup]);
        proj_k<<<B_ / 8, 256, 0, stream>>>(
            h1b[oup], W_proj, b_proj, out, prev_y, t);
    }
}
